// MultimodalHead_20418274525662
// MI455X (gfx1250) — matrix operand order repaired and checked by reading, not compiled, not run on hardware
//
#include <hip/hip_runtime.h>
#include <math.h>

// ---------------------------------------------------------------------------
// MultimodalHead forward for MI455X (gfx1250, wave32, WMMA).
//
// Shapes: B=128, DIM=2048, S=256, P=16, NV=16.
// All GEMMs run as bf16 WMMA (V_WMMA_F32_16X16X32_BF16) with fp32 accum;
// LN / softmax / sigmoid epilogues absorb the bf16 input rounding.
// ---------------------------------------------------------------------------

#define BATCH 128
#define DIM   2048
#define IMGS  256
#define PP    16
#define NV    16

typedef __attribute__((ext_vector_type(16))) __bf16 v16bf;
typedef __attribute__((ext_vector_type(8)))  __bf16 v8bf;
typedef __attribute__((ext_vector_type(8)))  float  v8f;

// ---------------------------------------------------------------------------
// Generic GEMM: C[M=128, N] = A[128, K] * W[K, N] + bias, row-major fp32 in,
// fp32 out. Block tile 128(M) x 64(N) (full M per block), K-stage 64
// (= two 32-deep WMMA chunks per barrier). 256 threads = 8 waves arranged
// 4(M) x 2(N); each wave owns a 32x32 output tile with FOUR fp32
// accumulators, so every staged (A,B) fragment pair feeds 2 WMMAs
// (8 WMMAs per wave per stage) -- halves LDS reads per matrix op.
// A/B staged through LDS as bf16 (converted on store); LDS rows padded by
// 8 bf16 (144B = 9x16B) to keep 16B vector alignment while rotating banks.
// The FULL template removes all N-guards from interior blocks so the hot
// streaming loop is pure global_load -> v_cvt_pk_bf16 -> ds_store_b128.
// K must be a multiple of 64 (true for 2048/8192/4096/1024).
// ---------------------------------------------------------------------------
#define BM 128
#define BN 64
#define BK 64
#define LDP (BK + 8)

template <bool FULL>
__device__ __forceinline__ void gemm_body(const float* __restrict__ A,
                                          const float* __restrict__ W,
                                          const float* __restrict__ bias,
                                          float* __restrict__ C,
                                          int K, int N, int n0,
                                          __bf16 (*As)[LDP], __bf16 (*Bs)[LDP]) {
    const int tid  = threadIdx.x;
    const int lane = tid & 31;
    const int wave = tid >> 5;
    const int wm   = wave >> 1;     // 0..3  (32-row strip inside block)
    const int wn   = wave & 1;      // 0..1  (32-col strip inside block)
    const int half = lane >> 4;     // ISA lane-half (K interleave)
    const int mr   = lane & 15;

    // B staging assignment: one column per thread, 16 K-rows per thread.
    const int bn   = tid & 63;
    const int bkb  = (tid >> 6) * 16;
    const int gn   = n0 + bn;
    const bool okn = FULL || (gn < N);

    v8f c00 = {}, c01 = {}, c10 = {}, c11 = {};

    for (int k0 = 0; k0 < K; k0 += BK) {
        // ---- stage A tile (128 x 64): 32 consecutive floats per thread ----
        {
            const int row = tid >> 1;
            const int col = (tid & 1) * 32;
            const float4* src =
                (const float4*)(A + (size_t)row * K + k0 + col);
#pragma unroll
            for (int q = 0; q < 4; ++q) {
                const float4 f0 = src[2 * q];
                const float4 f1 = src[2 * q + 1];
                v8bf p;
                p[0] = (__bf16)f0.x; p[1] = (__bf16)f0.y;
                p[2] = (__bf16)f0.z; p[3] = (__bf16)f0.w;
                p[4] = (__bf16)f1.x; p[5] = (__bf16)f1.y;
                p[6] = (__bf16)f1.z; p[7] = (__bf16)f1.w;
                *(v8bf*)(&As[row][col + q * 8]) = p;
            }
        }
        // ---- stage B tile (64 x 64) transposed into Bs[n][k] --------------
        // Global reads coalesce across lanes (consecutive n at fixed k);
        // LDS writes are contiguous 16B vectors per thread.
        {
#pragma unroll
            for (int q = 0; q < 2; ++q) {
                v8bf p;
#pragma unroll
                for (int j = 0; j < 8; ++j) {
                    const int kk = bkb + q * 8 + j;
                    const float v =
                        okn ? W[(size_t)(k0 + kk) * N + gn] : 0.0f;
                    p[j] = (__bf16)v;
                }
                *(v8bf*)(&Bs[bn][bkb + q * 8]) = p;
            }
        }
        __syncthreads();

        // ---- two 32-deep WMMA chunks per stage; 4 frags -> 4 WMMAs each ---
#pragma unroll
        for (int kc = 0; kc < BK; kc += 32) {
            v16bf a0, a1;
#pragma unroll
            for (int e = 0; e < 16; ++e) {
                const int kk = kc + (e & 7) + half * 8 + (e >> 3) * 16;
                a0[e] = As[wm * 32 + mr][kk];
                a1[e] = As[wm * 32 + 16 + mr][kk];
            }
            v16bf b0, b1;
#pragma unroll
            for (int e = 0; e < 16; ++e) {
                const int kk = kc + (e & 7) + half * 8 + (e >> 3) * 16;
                b0[e] = Bs[wn * 32 + mr][kk];
                b1[e] = Bs[wn * 32 + 16 + mr][kk];
            }
            c00 = __builtin_amdgcn_wmma_f32_16x16x32_bf16(
                false, a0, false, b0, (short)0, c00, false, false);
            c01 = __builtin_amdgcn_wmma_f32_16x16x32_bf16(
                false, a0, false, b1, (short)0, c01, false, false);
            c10 = __builtin_amdgcn_wmma_f32_16x16x32_bf16(
                false, a1, false, b0, (short)0, c10, false, false);
            c11 = __builtin_amdgcn_wmma_f32_16x16x32_bf16(
                false, a1, false, b1, (short)0, c11, false, false);
        }
        __syncthreads();
    }

    // ---- epilogue: C layout (VGPR r, lane L): M = r + 8*(L>=16), N = L%16 -
    const int row0 = wm * 32 + half * 8;        // rows for c00 / c01
    const int row1 = row0 + 16;                 // rows for c10 / c11
    const int col0 = n0 + wn * 32 + mr;
    const int col1 = col0 + 16;
    if (FULL) {
        const float bv0 = bias[col0];
        const float bv1 = bias[col1];
#pragma unroll
        for (int r = 0; r < 8; ++r) {
            C[(size_t)(row0 + r) * N + col0] = c00[r] + bv0;
            C[(size_t)(row0 + r) * N + col1] = c01[r] + bv1;
            C[(size_t)(row1 + r) * N + col0] = c10[r] + bv0;
            C[(size_t)(row1 + r) * N + col1] = c11[r] + bv1;
        }
    } else {
        const float bv0 = (col0 < N) ? bias[col0] : 0.0f;
        const float bv1 = (col1 < N) ? bias[col1] : 0.0f;
#pragma unroll
        for (int r = 0; r < 8; ++r) {
            if (col0 < N) {
                C[(size_t)(row0 + r) * N + col0] = c00[r] + bv0;
                C[(size_t)(row1 + r) * N + col0] = c10[r] + bv0;
            }
            if (col1 < N) {
                C[(size_t)(row0 + r) * N + col1] = c01[r] + bv1;
                C[(size_t)(row1 + r) * N + col1] = c11[r] + bv1;
            }
        }
    }
}

__global__ __launch_bounds__(256)
void gemm_bias_wmma(const float* __restrict__ A, const float* __restrict__ W,
                    const float* __restrict__ bias, float* __restrict__ C,
                    int K, int N) {
    __shared__ __bf16 As[BM][LDP];
    __shared__ __bf16 Bs[BN][LDP];
    const int n0 = blockIdx.x * BN;
    if (n0 + BN <= N)
        gemm_body<true>(A, W, bias, C, K, N, n0, As, Bs);
    else
        gemm_body<false>(A, W, bias, C, K, N, n0, As, Bs);
}

// ---------------------------------------------------------------------------
// In-place LayerNorm + exact GELU per row. One 256-thread block per row.
// ---------------------------------------------------------------------------
__global__ __launch_bounds__(256)
void ln_gelu_kernel(float* __restrict__ H, const float* __restrict__ g,
                    const float* __restrict__ be, int Nf) {
    __shared__ float red[256];
    const int tid = threadIdx.x;
    float* h = H + (size_t)blockIdx.x * Nf;

    float s = 0.0f;
    for (int i = tid; i < Nf; i += 256) s += h[i];
    red[tid] = s; __syncthreads();
    for (int off = 128; off > 0; off >>= 1) {
        if (tid < off) red[tid] += red[tid + off];
        __syncthreads();
    }
    const float mean = red[0] / (float)Nf;
    __syncthreads();

    float v = 0.0f;
    for (int i = tid; i < Nf; i += 256) { float d = h[i] - mean; v += d * d; }
    red[tid] = v; __syncthreads();
    for (int off = 128; off > 0; off >>= 1) {
        if (tid < off) red[tid] += red[tid + off];
        __syncthreads();
    }
    const float inv = rsqrtf(red[0] / (float)Nf + 1e-5f);

    for (int i = tid; i < Nf; i += 256) {
        float t = (h[i] - mean) * inv * g[i] + be[i];
        h[i] = 0.5f * t * (1.0f + erff(t * 0.70710678118654752f));
    }
}

// ---------------------------------------------------------------------------
// Bilinear 16x16 -> 256x256 (half-pixel centers, clamped) + sigmoid.
// One thread per output pixel of [B,3,256,256].
// ---------------------------------------------------------------------------
__global__ __launch_bounds__(256)
void resize_sigmoid_kernel(const float* __restrict__ feats, float* __restrict__ out) {
    const unsigned idx = blockIdx.x * 256u + threadIdx.x;
    const unsigned total = (unsigned)BATCH * 3u * IMGS * IMGS;
    if (idx >= total) return;
    const int x  = idx & 255;
    const int y  = (idx >> 8) & 255;
    const int cb = idx >> 16;          // b*3 + c
    const int c  = cb % 3;
    const int b  = cb / 3;

    float sx = (x + 0.5f) * (float)PP / (float)IMGS - 0.5f;
    float sy = (y + 0.5f) * (float)PP / (float)IMGS - 0.5f;
    sx = fminf(fmaxf(sx, 0.0f), (float)(PP - 1));
    sy = fminf(fmaxf(sy, 0.0f), (float)(PP - 1));
    const int x0 = (int)sx, y0 = (int)sy;
    const int x1 = min(x0 + 1, PP - 1), y1 = min(y0 + 1, PP - 1);
    const float fx = sx - (float)x0, fy = sy - (float)y0;

    const float* f = feats + ((size_t)b * 3 + c) * PP * PP;
    const float v00 = f[y0 * PP + x0], v01 = f[y0 * PP + x1];
    const float v10 = f[y1 * PP + x0], v11 = f[y1 * PP + x1];
    const float v = (v00 * (1.f - fx) + v01 * fx) * (1.f - fy) +
                    (v10 * (1.f - fx) + v11 * fx) * fy;
    out[idx] = 1.0f / (1.0f + expf(-v));
}

// ---------------------------------------------------------------------------
// Softmax over last dim of [B, 14, 3].
// ---------------------------------------------------------------------------
__global__ __launch_bounds__(256)
void softmax3_kernel(const float* __restrict__ in, float* __restrict__ out) {
    const int idx = blockIdx.x * 256 + threadIdx.x;
    if (idx >= BATCH * (NV - 2)) return;
    const float a = in[idx * 3 + 0], b = in[idx * 3 + 1], c = in[idx * 3 + 2];
    const float m = fmaxf(a, fmaxf(b, c));
    const float ea = expf(a - m), eb = expf(b - m), ec = expf(c - m);
    const float inv = 1.0f / (ea + eb + ec);
    out[idx * 3 + 0] = ea * inv;
    out[idx * 3 + 1] = eb * inv;
    out[idx * 3 + 2] = ec * inv;
}

// ---------------------------------------------------------------------------
// Per-batch: orthogonal polar factor of 3x3 (== u @ vh from SVD) via Newton
// iteration X <- 0.5*(X + X^-T), then rotated[n,j] = sum_k vert[n,k]*R[j,k].
// ---------------------------------------------------------------------------
__global__ __launch_bounds__(128)
void polar_rotate_kernel(const float* __restrict__ raw,
                         const float* __restrict__ verts,
                         float* __restrict__ Rout,
                         float* __restrict__ rotated) {
    const int b = blockIdx.x * 128 + threadIdx.x;
    if (b >= BATCH) return;
    float X[9];
#pragma unroll
    for (int i = 0; i < 9; ++i) X[i] = raw[b * 9 + i];

    for (int it = 0; it < 20; ++it) {
        // cofactor matrix (inv(X)^T = cof(X)/det)
        float c00 =  X[4]*X[8] - X[5]*X[7];
        float c01 = -(X[3]*X[8] - X[5]*X[6]);
        float c02 =  X[3]*X[7] - X[4]*X[6];
        float c10 = -(X[1]*X[8] - X[2]*X[7]);
        float c11 =  X[0]*X[8] - X[2]*X[6];
        float c12 = -(X[0]*X[7] - X[1]*X[6]);
        float c20 =  X[1]*X[5] - X[2]*X[4];
        float c21 = -(X[0]*X[5] - X[2]*X[3]);
        float c22 =  X[0]*X[4] - X[1]*X[3];
        float det = X[0]*c00 + X[1]*c01 + X[2]*c02;
        if (fabsf(det) < 1e-20f) break;
        const float id = 1.0f / det;
        X[0] = 0.5f*(X[0] + c00*id); X[1] = 0.5f*(X[1] + c01*id); X[2] = 0.5f*(X[2] + c02*id);
        X[3] = 0.5f*(X[3] + c10*id); X[4] = 0.5f*(X[4] + c11*id); X[5] = 0.5f*(X[5] + c12*id);
        X[6] = 0.5f*(X[6] + c20*id); X[7] = 0.5f*(X[7] + c21*id); X[8] = 0.5f*(X[8] + c22*id);
    }
#pragma unroll
    for (int i = 0; i < 9; ++i) Rout[b * 9 + i] = X[i];

    const float* v = verts + (size_t)b * NV * 3;
    float* ro = rotated + (size_t)b * NV * 3;
    for (int n = 0; n < NV; ++n) {
        const float vx = v[n * 3 + 0], vy = v[n * 3 + 1], vz = v[n * 3 + 2];
#pragma unroll
        for (int j = 0; j < 3; ++j)
            ro[n * 3 + j] = vx * X[j * 3 + 0] + vy * X[j * 3 + 1] + vz * X[j * 3 + 2];
    }
}

// ---------------------------------------------------------------------------
// Zero-fill (float4 stores) for the two scatter-rendered image buffers.
// ---------------------------------------------------------------------------
__global__ __launch_bounds__(256)
void zero4_kernel(float4* __restrict__ p, unsigned n4) {
    const unsigned i = blockIdx.x * 256u + threadIdx.x;
    if (i < n4) p[i] = make_float4(0.f, 0.f, 0.f, 0.f);
}

// ---------------------------------------------------------------------------
// Scatter render: one thread per (batch, vertex); writes 1.0 into both images
// (original + rotated mesh). Races are benign (all writers store 1.0); this
// matches the reference's .max() semantics since invalid verts never write.
// ---------------------------------------------------------------------------
__global__ __launch_bounds__(256)
void scatter_kernel(const float* __restrict__ verts, const float* __restrict__ rot,
                    float* __restrict__ imgO, float* __restrict__ imgR) {
    const int idx = blockIdx.x * 256 + threadIdx.x;
    if (idx >= BATCH * NV) return;
    const int b = idx / NV;
    const float* v0 = verts + (size_t)idx * 3;
    const float* v1 = rot   + (size_t)idx * 3;
    const size_t base = (size_t)b * 3 * IMGS * IMGS;

    {
        const int xi = (int)((v0[0] + 1.0f) * (IMGS * 0.5f));
        const int yi = (int)((v0[1] + 1.0f) * (IMGS * 0.5f));
        if (xi >= 0 && xi < IMGS && yi >= 0 && yi < IMGS) {
            const size_t o = base + (size_t)yi * IMGS + xi;
            imgO[o] = 1.0f;
            imgO[o + (size_t)IMGS * IMGS] = 1.0f;
            imgO[o + 2 * (size_t)IMGS * IMGS] = 1.0f;
        }
    }
    {
        const int xi = (int)((v1[0] + 1.0f) * (IMGS * 0.5f));
        const int yi = (int)((v1[1] + 1.0f) * (IMGS * 0.5f));
        if (xi >= 0 && xi < IMGS && yi >= 0 && yi < IMGS) {
            const size_t o = base + (size_t)yi * IMGS + xi;
            imgR[o] = 1.0f;
            imgR[o + (size_t)IMGS * IMGS] = 1.0f;
            imgR[o + 2 * (size_t)IMGS * IMGS] = 1.0f;
        }
    }
}

// ---------------------------------------------------------------------------
// Host-side orchestration.
// ---------------------------------------------------------------------------
extern "C" void kernel_launch(void* const* d_in, const int* in_sizes, int n_in,
                              void* d_out, int out_size, void* d_ws, size_t ws_size,
                              hipStream_t stream) {
    const float* x     = (const float*)d_in[0];
    const float* tb_w  = (const float*)d_in[1];
    const float* tb_b  = (const float*)d_in[2];
    const float* ih_w1 = (const float*)d_in[3];
    const float* ih_b1 = (const float*)d_in[4];
    const float* ih_g  = (const float*)d_in[5];
    const float* ih_be = (const float*)d_in[6];
    const float* ih_w2 = (const float*)d_in[7];
    const float* ih_b2 = (const float*)d_in[8];
    const float* vg_w1 = (const float*)d_in[9];
    const float* vg_b1 = (const float*)d_in[10];
    const float* vg_g  = (const float*)d_in[11];
    const float* vg_be = (const float*)d_in[12];
    const float* vg_w2 = (const float*)d_in[13];
    const float* vg_b2 = (const float*)d_in[14];
    const float* fg_w1 = (const float*)d_in[15];
    const float* fg_b1 = (const float*)d_in[16];
    const float* fg_g  = (const float*)d_in[17];
    const float* fg_be = (const float*)d_in[18];
    const float* fg_w2 = (const float*)d_in[19];
    const float* fg_b2 = (const float*)d_in[20];
    const float* rp_w1 = (const float*)d_in[21];
    const float* rp_b1 = (const float*)d_in[22];
    const float* rp_g  = (const float*)d_in[23];
    const float* rp_be = (const float*)d_in[24];
    const float* rp_w2 = (const float*)d_in[25];
    const float* rp_b2 = (const float*)d_in[26];

    float* out = (float*)d_out;
    // d_out layout (flat floats, return order):
    const size_t off_text  = 0;                                             // 128*256
    const size_t off_image = off_text  + (size_t)BATCH*256;                 // 25,165,824
    const size_t off_vert  = off_image + (size_t)BATCH*3*IMGS*IMGS;         // 6144
    const size_t off_faces = off_vert  + (size_t)BATCH*NV*3;                // 5376
    const size_t off_R     = off_faces + (size_t)BATCH*(NV-2)*3;            // 1152
    const size_t off_rot   = off_R     + (size_t)BATCH*9;                   // 6144
    const size_t off_imgO  = off_rot   + (size_t)BATCH*NV*3;
    const size_t off_imgR  = off_imgO  + (size_t)BATCH*3*IMGS*IMGS;

    // workspace layout (floats)
    float* W = (float*)d_ws;
    float* h_img   = W;                             // 128*8192
    float* feats   = h_img  + (size_t)BATCH*8192;   // 128*768
    float* h_vg    = feats  + (size_t)BATCH*768;    // 128*4096
    float* h_fg    = h_vg   + (size_t)BATCH*4096;   // 128*4096
    float* h_rp    = h_fg   + (size_t)BATCH*4096;   // 128*1024
    float* fg_lgt  = h_rp   + (size_t)BATCH*1024;   // 128*42
    float* rp_raw  = fg_lgt + (size_t)BATCH*42;     // 128*9

    dim3 blk(256);
    auto gemm_grid = [](int N) { return dim3((unsigned)((N + BN - 1) / BN), 1); };

    // text head
    gemm_bias_wmma<<<gemm_grid(256), blk, 0, stream>>>(x, tb_w, tb_b, out + off_text, DIM, 256);

    // image head
    gemm_bias_wmma<<<gemm_grid(8192), blk, 0, stream>>>(x, ih_w1, ih_b1, h_img, DIM, 8192);
    ln_gelu_kernel<<<BATCH, blk, 0, stream>>>(h_img, ih_g, ih_be, 8192);
    gemm_bias_wmma<<<gemm_grid(768), blk, 0, stream>>>(h_img, ih_w2, ih_b2, feats, 8192, 768);
    {
        const unsigned total = (unsigned)BATCH * 3u * IMGS * IMGS;
        resize_sigmoid_kernel<<<(total + 255) / 256, blk, 0, stream>>>(feats, out + off_image);
    }

    // vertices
    gemm_bias_wmma<<<gemm_grid(4096), blk, 0, stream>>>(x, vg_w1, vg_b1, h_vg, DIM, 4096);
    ln_gelu_kernel<<<BATCH, blk, 0, stream>>>(h_vg, vg_g, vg_be, 4096);
    gemm_bias_wmma<<<gemm_grid(48), blk, 0, stream>>>(h_vg, vg_w2, vg_b2, out + off_vert, 4096, 48);

    // faces
    gemm_bias_wmma<<<gemm_grid(4096), blk, 0, stream>>>(x, fg_w1, fg_b1, h_fg, DIM, 4096);
    ln_gelu_kernel<<<BATCH, blk, 0, stream>>>(h_fg, fg_g, fg_be, 4096);
    gemm_bias_wmma<<<gemm_grid(42), blk, 0, stream>>>(h_fg, fg_w2, fg_b2, fg_lgt, 4096, 42);
    softmax3_kernel<<<(BATCH * (NV - 2) + 255) / 256, blk, 0, stream>>>(fg_lgt, out + off_faces);

    // rotation
    gemm_bias_wmma<<<gemm_grid(1024), blk, 0, stream>>>(x, rp_w1, rp_b1, h_rp, DIM, 1024);
    ln_gelu_kernel<<<BATCH, blk, 0, stream>>>(h_rp, rp_g, rp_be, 1024);
    gemm_bias_wmma<<<gemm_grid(9), blk, 0, stream>>>(h_rp, rp_w2, rp_b2, rp_raw, 1024, 9);
    polar_rotate_kernel<<<1, 128, 0, stream>>>(rp_raw, out + off_vert, out + off_R, out + off_rot);

    // renders (zero both images, then scatter both meshes)
    {
        const unsigned n4 = (unsigned)(2ull * BATCH * 3 * IMGS * IMGS / 4);
        zero4_kernel<<<(n4 + 255) / 256, blk, 0, stream>>>((float4*)(out + off_imgO), n4);
        scatter_kernel<<<(BATCH * NV + 255) / 256, blk, 0, stream>>>(
            out + off_vert, out + off_rot, out + off_imgO, out + off_imgR);
    }
}
